// STPSPPWithGridCNN_62654982914311
// MI455X (gfx1250) — matrix operand order repaired and checked by reading, not compiled, not run on hardware
//
#include <hip/hip_runtime.h>
#include <hip/hip_bf16.h>
#include <cstdint>
#include <cstddef>

// ---------------------------------------------------------------------------
// Types / helpers
// ---------------------------------------------------------------------------
typedef _Float16 half_t;
typedef __attribute__((ext_vector_type(16))) _Float16 v16h;
typedef __attribute__((ext_vector_type(8)))  float    v8f;

#define CDIV(a, b) (((a) + (b) - 1) / (b))

__device__ __forceinline__ float sigm(float x) { return 1.0f / (1.0f + __expf(-x)); }

union AFrag { v16h h; uint32_t u[8]; };

// ---------------------------------------------------------------------------
// WMMA GEMM:  C[M,N] (f32, ldc) = A[M,K](f16,lda) * B[N,K]^T(f16,ldb)
//             + (accum ? C : 0) + bias[n]
// M % 16 == 0, N % 16 == 0, K % 32 == 0.  One wave per 16x16 C tile.
// ---------------------------------------------------------------------------
__global__ void __launch_bounds__(128)
k_gemm_wmma(const half_t* __restrict__ A, int lda,
            const half_t* __restrict__ B, int ldb,
            float* __restrict__ C, int ldc,
            const float* __restrict__ bias,
            int M, int N, int K, int accum)
{
    int wave = blockIdx.x * (blockDim.x >> 5) + (threadIdx.x >> 5);
    int ntN = N >> 4;
    int ntM = M >> 4;
    if (wave >= ntM * ntN) return;
    int mt = wave / ntN, nt = wave % ntN;
    int lane = threadIdx.x & 31;
    int l15 = lane & 15, hi = lane >> 4;
    int m = (mt << 4) + l15;   // A row for this lane's A-fragment
    int n = (nt << 4) + l15;   // B row (= C column) for this lane

    v8f acc;
    if (accum) {
#pragma unroll
        for (int v = 0; v < 8; ++v)
            acc[v] = C[(size_t)((mt << 4) + v + (hi << 3)) * ldc + n];
    } else {
#pragma unroll
        for (int v = 0; v < 8; ++v) acc[v] = 0.0f;
    }

    const uint32_t* Arow = (const uint32_t*)(A + (size_t)m * lda);
    const uint4*    Bq   = (const uint4*)(B + (size_t)n * ldb);

    for (int k0 = 0; k0 < K; k0 += 32) {
        AFrag a, b;
        // A fragment: VGPR v<4 -> K = k0 + 2v + 8*hi (+0,1) ; v>=4 -> +16
#pragma unroll
        for (int v = 0; v < 4; ++v)
            a.u[v] = Arow[(k0 + 2 * v + 8 * hi) >> 1];
#pragma unroll
        for (int v = 0; v < 4; ++v)
            a.u[4 + v] = Arow[(k0 + 16 + 2 * v + 8 * hi) >> 1];
        // B fragment: 8 halves at K = k0 + 8*hi, then 8 halves at K = k0 + 16 + 8*hi for row n
        uint4 w0 = Bq[(k0 >> 3) + hi];
        uint4 w1 = Bq[(k0 >> 3) + 2 + hi];
        b.u[0] = w0.x; b.u[1] = w0.y; b.u[2] = w0.z; b.u[3] = w0.w;
        b.u[4] = w1.x; b.u[5] = w1.y; b.u[6] = w1.z; b.u[7] = w1.w;

        acc = __builtin_amdgcn_wmma_f32_16x16x32_f16(
            false, a.h, false, b.h, (short)0, acc, false, false);
    }

    float bn = bias ? bias[n] : 0.0f;
#pragma unroll
    for (int v = 0; v < 8; ++v)
        C[(size_t)((mt << 4) + v + (hi << 3)) * ldc + n] = acc[v] + bn;
}

// ---------------------------------------------------------------------------
// LSTM gate pointwise.  z: (Mb, 4H) gate pre-activations (i,f,g,o).
// c: (Mb,H) state.  h16: (Mb,Hpad) f16 hidden for next-step GEMM.
// out[out_off + m*st_m + h]:  mode 0 store (opt tanh), 1 +=, 2 max.
// ---------------------------------------------------------------------------
__global__ void k_lstm_gates(const float* __restrict__ z, float* __restrict__ c,
                             half_t* __restrict__ h16, int Hpad, int Mb, int H,
                             float* __restrict__ out, long out_off, long st_m,
                             int mode, int tanh_store)
{
    long i = (long)blockIdx.x * blockDim.x + threadIdx.x;
    if (i >= (long)Mb * H) return;
    int m = (int)(i / H);
    int h = (int)(i % H);
    const float* zr = z + (size_t)m * 4 * H;
    float iv = sigm(zr[h]);
    float fv = sigm(zr[H + h]);
    float gv = tanhf(zr[2 * H + h]);
    float ov = sigm(zr[3 * H + h]);
    float cv = fv * c[i] + iv * gv;
    c[i] = cv;
    float hv = ov * tanhf(cv);
    h16[(size_t)m * Hpad + h] = (half_t)hv;
    float* op = out + out_off + (long)m * st_m + h;
    if (mode == 0)      *op = tanh_store ? tanhf(hv) : hv;
    else if (mode == 1) *op += hv;
    else                *op = fmaxf(*op, hv);
}

// ---------------------------------------------------------------------------
// Small utility kernels
// ---------------------------------------------------------------------------
__global__ void k_fill_f32(float* p, float v, long n) {
    long i = (long)blockIdx.x * blockDim.x + threadIdx.x;
    if (i < n) p[i] = v;
}
__global__ void k_fill_f16(half_t* p, long n) {
    long i = (long)blockIdx.x * blockDim.x + threadIdx.x;
    if (i < n) p[i] = (half_t)0.0f;
}
// dst[r,cPad] = (c<cols)? src[r,cols] : 0 ; also serves as pure f32->f16 convert
__global__ void k_cvt_pad_f16(const float* __restrict__ src, half_t* __restrict__ dst,
                              long rows, int cols, int colsPad) {
    long i = (long)blockIdx.x * blockDim.x + threadIdx.x;
    long n = rows * colsPad;
    if (i >= n) return;
    long r = i / colsPad;
    int  c = (int)(i % colsPad);
    dst[i] = (half_t)((c < cols) ? src[r * cols + c] : 0.0f);
}
__global__ void k_pad_rows_f16(const float* __restrict__ src, half_t* __restrict__ dst,
                               int R, int Rpad, int C) {
    long i = (long)blockIdx.x * blockDim.x + threadIdx.x;
    long n = (long)Rpad * C;
    if (i >= n) return;
    int r = (int)(i / C);
    dst[i] = (half_t)((r < R) ? src[i] : 0.0f);
}
__global__ void k_pad_vec_f32(const float* src, float* dst, int n, int npad) {
    int i = blockIdx.x * blockDim.x + threadIdx.x;
    if (i < npad) dst[i] = (i < n) ? src[i] : 0.0f;
}
__global__ void k_add2(const float* a, const float* b, float* o, int n) {
    int i = blockIdx.x * blockDim.x + threadIdx.x;
    if (i < n) o[i] = a[i] + b[i];
}
__global__ void k_tanh(const float* a, float* o, long n) {
    long i = (long)blockIdx.x * blockDim.x + threadIdx.x;
    if (i < n) o[i] = tanhf(a[i]);
}

// documents (16,40,30,300) -> xs16 (30, 640, 320) f16 zero-padded
__global__ void k_doc_to_x16(const float* __restrict__ doc, half_t* __restrict__ dst) {
    long i = (long)blockIdx.x * blockDim.x + threadIdx.x;
    long n = 30L * 640 * 320;
    if (i >= n) return;
    int t = (int)(i / (640 * 320));
    int r = (int)(i % (640 * 320));
    int m = r / 320, k = r % 320;
    float v = (k < 300) ? doc[((size_t)m * 30 + t) * 300 + k] : 0.0f;
    dst[i] = (half_t)v;
}

// sentpres = tanh(hsum/30): hsum (2,640,128) -> sp/sp16 (640, 256)
__global__ void k_sentpres(const float* __restrict__ hsum, float* __restrict__ sp,
                           half_t* __restrict__ sp16) {
    long i = (long)blockIdx.x * blockDim.x + threadIdx.x;
    if (i >= 640L * 256) return;
    int m = (int)(i / 256), c = (int)(i % 256);
    int d = c / 128, h = c % 128;
    float v = tanhf(hsum[(size_t)d * 640 * 128 + (size_t)m * 128 + h] * (1.0f / 30.0f));
    sp[i] = v;
    sp16[i] = (half_t)v;
}

// tag input (dl,B,256) f16 from sentpres (B,dl,256)
__global__ void k_tag_in16(const float* __restrict__ sp, half_t* __restrict__ dst) {
    long i = (long)blockIdx.x * blockDim.x + threadIdx.x;
    if (i >= 40L * 16 * 256) return;
    int k = (int)(i % 256);
    int b = (int)((i / 256) % 16);
    int t = (int)(i / (16 * 256));
    dst[i] = (half_t)sp[((size_t)b * 40 + t) * 256 + k];
}

// feat[b,l,m] = tanh( dot(k[b,l,:], q[b,m,:]) )
__global__ void k_feat(const float* __restrict__ kb, const float* __restrict__ qb,
                       float* __restrict__ feat, int d) {
    int i = blockIdx.x * blockDim.x + threadIdx.x;
    if (i >= 16 * 40 * 40) return;
    int m = i % 40, l = (i / 40) % 40, b = i / 1600;
    const float* kr = kb + ((size_t)b * 40 + l) * d;
    const float* qr = qb + ((size_t)b * 40 + m) * d;
    float s = 0.0f;
    for (int j = 0; j < d; ++j) s += kr[j] * qr[j];
    feat[i] = tanhf(s);
}

// adaptive max pool bins (1,3,5,7) over last axis (L=40) -> (640,16)
__global__ void k_spp(const float* __restrict__ feat, float* __restrict__ out) {
    int i = blockIdx.x * blockDim.x + threadIdx.x;
    if (i >= 640) return;
    const float* fr = feat + (size_t)i * 40;
    float* orow = out + (size_t)i * 16;
    const int sizes[4] = {1, 3, 5, 7};
    int col = 0;
    for (int si = 0; si < 4; ++si) {
        int s = sizes[si];
        for (int j = 0; j < s; ++j) {
            int lo = (j * 40) / s;
            int hiq = ((j + 1) * 40 + s - 1) / s;
            float mx = -1e30f;
            for (int t = lo; t < hiq; ++t) mx = fmaxf(mx, fr[t]);
            orow[col++] = mx;
        }
    }
}

// tag_cat = [tag_out(512) | sentFt(16) | roleFt(16)] -> (640,544) f32 + f16
__global__ void k_tagcat(const float* __restrict__ to, const float* __restrict__ sf,
                         const float* __restrict__ rf, float* __restrict__ cat,
                         half_t* __restrict__ cat16) {
    long i = (long)blockIdx.x * blockDim.x + threadIdx.x;
    if (i >= 640L * 544) return;
    int m = (int)(i / 544), c = (int)(i % 544);
    float v;
    if (c < 512)      v = to[(size_t)m * 512 + c];
    else if (c < 528) v = sf[(size_t)m * 16 + (c - 512)];
    else              v = rf[(size_t)m * 16 + (c - 528)];
    cat[i] = v;
    cat16[i] = (half_t)v;
}

__global__ void k_logsoftmax(const float* __restrict__ z, int ldz,
                             float* __restrict__ out, int R, int C) {
    int r = blockIdx.x * blockDim.x + threadIdx.x;
    if (r >= R) return;
    const float* zr = z + (size_t)r * ldz;
    float mx = zr[0];
    for (int c = 1; c < C; ++c) mx = fmaxf(mx, zr[c]);
    float s = 0.0f;
    for (int c = 0; c < C; ++c) s += __expf(zr[c] - mx);
    float lse = mx + __logf(s);
    for (int c = 0; c < C; ++c) out[(size_t)r * C + c] = zr[c] - lse;
}

// gm[b,i,j,:] = grid==0 ? 0 : tag_cat[b, grid[b,i,j], :]
__global__ void k_gm(const float* __restrict__ cat, const int* __restrict__ grid,
                     float* __restrict__ gm) {
    long i = (long)blockIdx.x * blockDim.x + threadIdx.x;
    long n = 16L * 400 * 544;
    if (i >= n) return;
    int d = (int)(i % 544);
    long r = i / 544;          // b*400 + i*20 + j
    int b = (int)(r / 400);
    int g = grid[r];
    gm[i] = (g == 0) ? 0.0f : cat[((size_t)b * 40 + g) * 544 + d];
}

// px16[t,m,k] = (f16) gm[(m*20 + t)*544 + k]     (m = b*20+i)
__global__ void k_px16(const float* __restrict__ gm, half_t* __restrict__ px) {
    long i = (long)blockIdx.x * blockDim.x + threadIdx.x;
    long n = 20L * 320 * 544;
    if (i >= n) return;
    int k = (int)(i % 544);
    int m = (int)((i / 544) % 320);
    int t = (int)(i / (320L * 544));
    px[i] = (half_t)gm[((size_t)m * 20 + t) * 544 + k];
}

// img[b,d,i,j] = j==0 ? paras_out[b*20+i, d] : gm[b,i,j-1,d]
__global__ void k_img(const float* __restrict__ pmax, const float* __restrict__ gm,
                      float* __restrict__ img) {
    long idx = (long)blockIdx.x * blockDim.x + threadIdx.x;
    long n = 16L * 544 * 400;
    if (idx >= n) return;
    int j = (int)(idx % 20);
    int i = (int)((idx / 20) % 20);
    int d = (int)((idx / 400) % 544);
    int b = (int)(idx / (544L * 400));
    float v;
    if (j == 0) v = pmax[((size_t)b * 20 + i) * 544 + d];
    else        v = gm[(((size_t)b * 20 + i) * 20 + (j - 1)) * 544 + d];
    img[idx] = v;
}

// 3x3 SAME conv + bias + relu.  in (16,Cin,S,S), W (Cout,Cin,3,3)
__global__ void k_conv_relu(const float* __restrict__ in, const float* __restrict__ Wt,
                            const float* __restrict__ bias, float* __restrict__ out,
                            int Cin, int Cout, int S) {
    long i = (long)blockIdx.x * blockDim.x + threadIdx.x;
    long n = 16L * Cout * S * S;
    if (i >= n) return;
    int x = (int)(i % S);
    int y = (int)((i / S) % S);
    int co = (int)((i / ((long)S * S)) % Cout);
    int b = (int)(i / ((long)S * S * Cout));
    float s = bias[co];
    for (int ci = 0; ci < Cin; ++ci) {
        const float* ip = in + ((size_t)b * Cin + ci) * S * S;
        const float* wp = Wt + ((size_t)co * Cin + ci) * 9;
        for (int ky = 0; ky < 3; ++ky) {
            int yy = y + ky - 1;
            if (yy < 0 || yy >= S) continue;
            for (int kx = 0; kx < 3; ++kx) {
                int xx = x + kx - 1;
                if (xx < 0 || xx >= S) continue;
                s += ip[yy * S + xx] * wp[ky * 3 + kx];
            }
        }
    }
    out[i] = fmaxf(s, 0.0f);
}

__global__ void k_pool2(const float* __restrict__ in, float* __restrict__ out,
                        int C, int Ho) {
    long i = (long)blockIdx.x * blockDim.x + threadIdx.x;
    long n = 16L * C * Ho * Ho;
    if (i >= n) return;
    int x = (int)(i % Ho);
    int y = (int)((i / Ho) % Ho);
    int c = (int)((i / ((long)Ho * Ho)) % C);
    int b = (int)(i / ((long)Ho * Ho * C));
    int S = 2 * Ho;
    const float* ip = in + ((size_t)b * C + c) * S * S;
    float m = fmaxf(fmaxf(ip[(2 * y) * S + 2 * x], ip[(2 * y) * S + 2 * x + 1]),
                    fmaxf(ip[(2 * y + 1) * S + 2 * x], ip[(2 * y + 1) * S + 2 * x + 1]));
    out[i] = m;
}

// out[r,c] = bias[c] + dot(A[r,:K], W[c,:K])
__global__ void k_linear_small(const float* __restrict__ A, const float* __restrict__ Wt,
                               const float* __restrict__ bias, float* __restrict__ out,
                               int M, int N, int K) {
    int i = blockIdx.x * blockDim.x + threadIdx.x;
    if (i >= M * N) return;
    int r = i / N, c = i % N;
    const float* ar = A + (size_t)r * K;
    const float* wr = Wt + (size_t)c * K;
    float s = bias[c];
    for (int k = 0; k < K; ++k) s += ar[k] * wr[k];
    out[i] = s;
}

// ---------------------------------------------------------------------------
// Host driver
// ---------------------------------------------------------------------------
extern "C" void kernel_launch(void* const* d_in, const int* in_sizes, int n_in,
                              void* d_out, int out_size, void* d_ws, size_t ws_size,
                              hipStream_t stream) {
    (void)in_sizes; (void)n_in; (void)out_size; (void)ws_size;
    const float* documents = (const float*)d_in[0];
    const float* sent_Wih = (const float*)d_in[2];
    const float* sent_Whh = (const float*)d_in[3];
    const float* sent_bih = (const float*)d_in[4];
    const float* sent_bhh = (const float*)d_in[5];
    const float* tag_Wih  = (const float*)d_in[6];
    const float* tag_Whh  = (const float*)d_in[7];
    const float* tag_bih  = (const float*)d_in[8];
    const float* tag_bhh  = (const float*)d_in[9];
    const float* para_Wih = (const float*)d_in[10];
    const float* para_Whh = (const float*)d_in[11];
    const float* para_bih = (const float*)d_in[12];
    const float* para_bhh = (const float*)d_in[13];
    const float* sf_Wk = (const float*)d_in[14];
    const float* sf_bk = (const float*)d_in[15];
    const float* sf_Wq = (const float*)d_in[16];
    const float* sf_bq = (const float*)d_in[17];
    const float* rf_Wk = (const float*)d_in[18];
    const float* rf_bk = (const float*)d_in[19];
    const float* rf_Wq = (const float*)d_in[20];
    const float* rf_bq = (const float*)d_in[21];
    const float* cls_W = (const float*)d_in[22];
    const float* cls_b = (const float*)d_in[23];
    const float* pcl_W = (const float*)d_in[24];
    const float* pcl_b = (const float*)d_in[25];
    const float* c1_W  = (const float*)d_in[26];
    const float* c1_b  = (const float*)d_in[27];
    const float* c2_W  = (const float*)d_in[28];
    const float* c2_b  = (const float*)d_in[29];
    const float* sc_W  = (const float*)d_in[30];
    const float* sc_b  = (const float*)d_in[31];
    const int*   grid  = (const int*)d_in[32];
    float* out = (float*)d_out;

    // ---- workspace carve-out -------------------------------------------
    char* Wb = (char*)d_ws;
    size_t off = 0;
    auto alloc = [&](size_t bytes) -> char* {
        off = (off + 255) & ~(size_t)255;
        char* p = Wb + off;
        off += bytes;
        return p;
    };
    half_t* xs16    = (half_t*)alloc(30L * 640 * 320 * 2);
    half_t* sWih16  = (half_t*)alloc(2L * 512 * 320 * 2);
    half_t* sWhh16  = (half_t*)alloc(2L * 512 * 128 * 2);
    float*  sBias   = (float*) alloc(2L * 512 * 4);
    half_t* tWih16  = (half_t*)alloc(2L * 1024 * 256 * 2);
    half_t* tWhh16  = (half_t*)alloc(2L * 1024 * 256 * 2);
    float*  tBias   = (float*) alloc(2L * 1024 * 4);
    half_t* pWih16  = (half_t*)alloc(2L * 1088 * 544 * 2);
    half_t* pWhh16  = (half_t*)alloc(2L * 1088 * 288 * 2);
    float*  pBias   = (float*) alloc(2L * 1088 * 4);
    half_t* sfWk16  = (half_t*)alloc(256L * 256 * 2);
    half_t* sfWq16  = (half_t*)alloc(256L * 256 * 2);
    half_t* rfWk16  = (half_t*)alloc(512L * 512 * 2);
    half_t* rfWq16  = (half_t*)alloc(512L * 512 * 2);
    half_t* clsW16  = (half_t*)alloc(16L * 544 * 2);
    float*  clsB    = (float*) alloc(16 * 4);
    half_t* pclW16  = (half_t*)alloc(16L * 544 * 2);
    float*  pclB    = (float*) alloc(16 * 4);
    float*  zbuf    = (float*) alloc(2L * 320 * 1088 * 4);  // >= all LSTM z sizes
    float*  cbuf    = (float*) alloc(2L * 320 * 288 * 4);
    half_t* hbuf    = (half_t*)alloc(2L * 320 * 288 * 2);
    float*  hsum    = (float*) alloc(2L * 640 * 128 * 4);
    float*  sp      = (float*) alloc(640L * 256 * 4);
    half_t* sp16    = (half_t*)alloc(640L * 256 * 2);
    half_t* tagin16 = (half_t*)alloc(40L * 16 * 256 * 2);
    float*  kbuf    = (float*) alloc(640L * 512 * 4);
    float*  qbuf    = (float*) alloc(640L * 512 * 4);
    float*  featb   = (float*) alloc(16L * 40 * 40 * 4);
    float*  sentFt  = (float*) alloc(640L * 16 * 4);
    float*  roleFt  = (float*) alloc(640L * 16 * 4);
    float*  tagout  = (float*) alloc(640L * 512 * 4);
    half_t* tagout16= (half_t*)alloc(640L * 512 * 2);
    float*  tagcat  = (float*) alloc(640L * 544 * 4);
    half_t* tagcat16= (half_t*)alloc(640L * 544 * 2);
    float*  logits  = (float*) alloc(640L * 16 * 4);
    float*  gmb     = (float*) alloc(16L * 400 * 544 * 4);
    half_t* px16    = (half_t*)alloc(20L * 320 * 544 * 2);
    float*  pmax    = (float*) alloc(320L * 544 * 4);
    half_t* pout16  = (half_t*)alloc(320L * 544 * 2);
    float*  img     = (float*) alloc(16L * 544 * 400 * 4);
    float*  c1o     = (float*) alloc(16L * 64 * 400 * 4);
    float*  p1      = (float*) alloc(16L * 64 * 100 * 4);
    float*  c2o     = (float*) alloc(16L * 32 * 100 * 4);
    float*  p2      = (float*) alloc(16L * 32 * 25 * 4);
    float*  thb     = (float*) alloc(16L * 800 * 4);
    float*  slog    = (float*) alloc(16L * 4 * 4);

    auto gemm = [&](const half_t* A, int lda, const half_t* Bm, int ldb,
                    float* C, int ldc, const float* bias,
                    int M, int N, int K, int accum) {
        int tiles = (M / 16) * (N / 16);
        k_gemm_wmma<<<CDIV(tiles, 4), 128, 0, stream>>>(A, lda, Bm, ldb, C, ldc,
                                                        bias, M, N, K, accum);
    };
    auto fillf = [&](float* p, float v, long n) {
        k_fill_f32<<<CDIV(n, 256), 256, 0, stream>>>(p, v, n);
    };
    auto fillh = [&](half_t* p, long n) {
        k_fill_f16<<<CDIV(n, 256), 256, 0, stream>>>(p, n);
    };

    // ---- weight / input conversion -------------------------------------
    k_doc_to_x16<<<CDIV(30L * 640 * 320, 256), 256, 0, stream>>>(documents, xs16);
    k_cvt_pad_f16<<<CDIV(1024L * 320, 256), 256, 0, stream>>>(sent_Wih, sWih16, 1024, 300, 320);
    k_cvt_pad_f16<<<CDIV(1024L * 128, 256), 256, 0, stream>>>(sent_Whh, sWhh16, 1024, 128, 128);
    k_add2<<<CDIV(1024, 256), 256, 0, stream>>>(sent_bih, sent_bhh, sBias, 1024);
    k_cvt_pad_f16<<<CDIV(2048L * 256, 256), 256, 0, stream>>>(tag_Wih, tWih16, 2048, 256, 256);
    k_cvt_pad_f16<<<CDIV(2048L * 256, 256), 256, 0, stream>>>(tag_Whh, tWhh16, 2048, 256, 256);
    k_add2<<<CDIV(2048, 256), 256, 0, stream>>>(tag_bih, tag_bhh, tBias, 2048);
    k_cvt_pad_f16<<<CDIV(2176L * 544, 256), 256, 0, stream>>>(para_Wih, pWih16, 2176, 544, 544);
    k_cvt_pad_f16<<<CDIV(2176L * 288, 256), 256, 0, stream>>>(para_Whh, pWhh16, 2176, 272, 288);
    k_add2<<<CDIV(2176, 256), 256, 0, stream>>>(para_bih, para_bhh, pBias, 2176);
    k_cvt_pad_f16<<<CDIV(256L * 256, 256), 256, 0, stream>>>(sf_Wk, sfWk16, 256, 256, 256);
    k_cvt_pad_f16<<<CDIV(256L * 256, 256), 256, 0, stream>>>(sf_Wq, sfWq16, 256, 256, 256);
    k_cvt_pad_f16<<<CDIV(512L * 512, 256), 256, 0, stream>>>(rf_Wk, rfWk16, 512, 512, 512);
    k_cvt_pad_f16<<<CDIV(512L * 512, 256), 256, 0, stream>>>(rf_Wq, rfWq16, 512, 512, 512);
    k_pad_rows_f16<<<CDIV(16L * 544, 256), 256, 0, stream>>>(cls_W, clsW16, 8, 16, 544);
    k_pad_vec_f32<<<1, 16, 0, stream>>>(cls_b, clsB, 8, 16);
    k_pad_rows_f16<<<CDIV(16L * 544, 256), 256, 0, stream>>>(pcl_W, pclW16, 8, 16, 544);
    k_pad_vec_f32<<<1, 16, 0, stream>>>(pcl_b, pclB, 8, 16);

    // ---- sentence BiLSTM (T=30, Mb=640, H=128) --------------------------
    fillh(hbuf, 2L * 640 * 128);
    fillf(cbuf, 0.0f, 2L * 640 * 128);
    fillf(hsum, 0.0f, 2L * 640 * 128);
    for (int t = 0; t < 30; ++t) {
        for (int d = 0; d < 2; ++d) {
            int te = d ? (29 - t) : t;
            const half_t* xt = xs16 + (size_t)te * 640 * 320;
            float* zd = zbuf + (size_t)d * 640 * 512;
            gemm(xt, 320, sWih16 + (size_t)d * 512 * 320, 320, zd, 512,
                 sBias + d * 512, 640, 512, 320, 0);
            gemm(hbuf + (size_t)d * 640 * 128, 128, sWhh16 + (size_t)d * 512 * 128, 128,
                 zd, 512, nullptr, 640, 512, 128, 1);
            k_lstm_gates<<<CDIV(640L * 128, 256), 256, 0, stream>>>(
                zd, cbuf + (size_t)d * 640 * 128, hbuf + (size_t)d * 640 * 128, 128,
                640, 128, hsum, (long)d * 640 * 128, 128, 1, 0);
        }
    }
    k_sentpres<<<CDIV(640L * 256, 256), 256, 0, stream>>>(hsum, sp, sp16);

    // ---- sentFt (SPP over sentpres, d=256) ------------------------------
    gemm(sp16, 256, sfWk16, 256, kbuf, 256, sf_bk, 640, 256, 256, 0);
    gemm(sp16, 256, sfWq16, 256, qbuf, 256, sf_bq, 640, 256, 256, 0);
    k_feat<<<CDIV(16 * 40 * 40, 256), 256, 0, stream>>>(kbuf, qbuf, featb, 256);
    k_spp<<<CDIV(640, 256), 256, 0, stream>>>(featb, sentFt);

    // ---- tag BiLSTM (T=40, Mb=16, H=256) --------------------------------
    k_tag_in16<<<CDIV(40L * 16 * 256, 256), 256, 0, stream>>>(sp, tagin16);
    fillh(hbuf, 2L * 16 * 256);
    fillf(cbuf, 0.0f, 2L * 16 * 256);
    for (int t = 0; t < 40; ++t) {
        for (int d = 0; d < 2; ++d) {
            int te = d ? (39 - t) : t;
            const half_t* xt = tagin16 + (size_t)te * 16 * 256;
            float* zd = zbuf + (size_t)d * 16 * 1024;
            gemm(xt, 256, tWih16 + (size_t)d * 1024 * 256, 256, zd, 1024,
                 tBias + d * 1024, 16, 1024, 256, 0);
            gemm(hbuf + (size_t)d * 16 * 256, 256, tWhh16 + (size_t)d * 1024 * 256, 256,
                 zd, 1024, nullptr, 16, 1024, 256, 1);
            // tag_out[b][te][d*256 + h] = tanh(h)
            k_lstm_gates<<<CDIV(16L * 256, 256), 256, 0, stream>>>(
                zd, cbuf + (size_t)d * 16 * 256, hbuf + (size_t)d * 16 * 256, 256,
                16, 256, tagout, (long)te * 512 + d * 256, 40L * 512, 0, 1);
        }
    }
    k_cvt_pad_f16<<<CDIV(640L * 512, 256), 256, 0, stream>>>(tagout, tagout16, 640, 512, 512);

    // ---- roleFt (SPP over tag_out, d=512) -------------------------------
    gemm(tagout16, 512, rfWk16, 512, kbuf, 512, rf_bk, 640, 512, 512, 0);
    gemm(tagout16, 512, rfWq16, 512, qbuf, 512, rf_bq, 640, 512, 512, 0);
    k_feat<<<CDIV(16 * 40 * 40, 256), 256, 0, stream>>>(kbuf, qbuf, featb, 512);
    k_spp<<<CDIV(640, 256), 256, 0, stream>>>(featb, roleFt);

    // ---- tag_cat + classifier (result) ----------------------------------
    k_tagcat<<<CDIV(640L * 544, 256), 256, 0, stream>>>(tagout, sentFt, roleFt,
                                                        tagcat, tagcat16);
    gemm(tagcat16, 544, clsW16, 544, logits, 16, clsB, 640, 16, 544, 0);
    k_logsoftmax<<<CDIV(640, 256), 256, 0, stream>>>(logits, 16, out, 640, 8);

    // ---- grid gather + para BiLSTM (T=20, Mb=320, H=272) ----------------
    k_gm<<<CDIV(16L * 400 * 544, 256), 256, 0, stream>>>(tagcat, grid, gmb);
    k_px16<<<CDIV(20L * 320 * 544, 256), 256, 0, stream>>>(gmb, px16);
    fillh(hbuf, 2L * 320 * 288);           // zero incl. K-pad columns
    fillf(cbuf, 0.0f, 2L * 320 * 272);
    fillf(pmax, -1e30f, 320L * 544);
    for (int t = 0; t < 20; ++t) {
        for (int d = 0; d < 2; ++d) {
            int te = d ? (19 - t) : t;
            const half_t* xt = px16 + (size_t)te * 320 * 544;
            float* zd = zbuf + (size_t)d * 320 * 1088;
            gemm(xt, 544, pWih16 + (size_t)d * 1088 * 544, 544, zd, 1088,
                 pBias + d * 1088, 320, 1088, 544, 0);
            gemm(hbuf + (size_t)d * 320 * 288, 288, pWhh16 + (size_t)d * 1088 * 288, 288,
                 zd, 1088, nullptr, 320, 1088, 288, 1);
            // running max over time: pmax[m][d*272 + h]
            k_lstm_gates<<<CDIV(320L * 272, 256), 256, 0, stream>>>(
                zd, cbuf + (size_t)d * 320 * 272, hbuf + (size_t)d * 320 * 288, 288,
                320, 272, pmax, (long)d * 272, 544, 2, 0);
        }
    }
    k_cvt_pad_f16<<<CDIV(320L * 544, 256), 256, 0, stream>>>(pmax, pout16, 320, 544, 544);

    // ---- paragraph classifier (paras) -----------------------------------
    gemm(pout16, 544, pclW16, 544, logits, 16, pclB, 320, 16, 544, 0);
    k_logsoftmax<<<CDIV(320, 256), 256, 0, stream>>>(logits, 16, out + 5184, 320, 8);

    // ---- grid CNN (scores) ----------------------------------------------
    k_img<<<CDIV(16L * 544 * 400, 256), 256, 0, stream>>>(pmax, gmb, img);
    k_conv_relu<<<CDIV(16L * 64 * 400, 256), 256, 0, stream>>>(img, c1_W, c1_b, c1o, 544, 64, 20);
    k_pool2<<<CDIV(16L * 64 * 100, 256), 256, 0, stream>>>(c1o, p1, 64, 10);
    k_conv_relu<<<CDIV(16L * 32 * 100, 256), 256, 0, stream>>>(p1, c2_W, c2_b, c2o, 64, 32, 10);
    k_pool2<<<CDIV(16L * 32 * 25, 256), 256, 0, stream>>>(c2o, p2, 32, 5);
    k_tanh<<<CDIV(16L * 800, 256), 256, 0, stream>>>(p2, thb, 16L * 800);
    k_linear_small<<<CDIV(64, 64), 64, 0, stream>>>(thb, sc_W, sc_b, slog, 16, 4, 800);
    k_logsoftmax<<<1, 16, 0, stream>>>(slog, 4, out + 5120, 16, 4);
}
